// GatedGNN_16870631539211
// MI455X (gfx1250) — matrix operand order repaired and checked by reading, not compiled, not run on hardware
//
#include <hip/hip_runtime.h>
#include <hip/hip_bf16.h>

typedef __bf16 v16bf __attribute__((ext_vector_type(16)));
typedef float  v8f   __attribute__((ext_vector_type(8)));

__device__ __forceinline__ unsigned short f2bf(float f) {
    unsigned u = __float_as_uint(f);
    unsigned r = u + 0x7FFFu + ((u >> 16) & 1u);   // round-to-nearest-even
    return (unsigned short)(r >> 16);
}

__device__ __forceinline__ float sigmoidf_(float x) { return 1.0f / (1.0f + expf(-x)); }

// order-preserving float<->uint encoding for atomic max
__device__ __forceinline__ unsigned encf(float f) {
    unsigned u = __float_as_uint(f);
    return (u & 0x80000000u) ? ~u : (u | 0x80000000u);
}
__device__ __forceinline__ float decf(unsigned e) {
    unsigned u = (e & 0x80000000u) ? (e ^ 0x80000000u) : ~e;
    return __uint_as_float(u);
}

// ---------------------------------------------------------------------------
// Conversion kernels (hoist all fp32->bf16 work out of the GEMM hot loop)
// ---------------------------------------------------------------------------
// out[r][kp] (ld=Kpad) = kp<K ? bf16(in[r*K+kp]) : 0
__global__ void gg_cvt_pad(const float* __restrict__ in, unsigned short* __restrict__ out,
                           int M, int K, int Kpad) {
    unsigned idx = blockIdx.x * 256 + threadIdx.x;
    if (idx >= (unsigned)M * (unsigned)Kpad) return;
    int r = idx / Kpad, kp = idx % Kpad;
    out[idx] = (kp < K) ? f2bf(in[(size_t)r * K + kp]) : (unsigned short)0;
}

// transpose-convert: out[n][kp] (ld=Kpad) = kp<K ? bf16(in[kp*Nc + n]) : 0
__global__ void gg_cvt_T(const float* __restrict__ in, unsigned short* __restrict__ out,
                         int K, int Nc, int Kpad) {
    unsigned idx = blockIdx.x * 256 + threadIdx.x;
    if (idx >= (unsigned)Nc * (unsigned)Kpad) return;
    int n = idx / Kpad, kp = idx % Kpad;
    out[idx] = (kp < K) ? f2bf(in[(size_t)kp * Nc + n]) : (unsigned short)0;
}

// ---------------------------------------------------------------------------
// bf16 WMMA GEMM, no LDS: C[M,Nc] = act(A @ B^T + bias)
//   A  : bf16 row-major [M][Kpad]          (Kpad multiple of 32, zero padded)
//   Bt : bf16 n-major   [Nc][Kpad]         (i.e. logical B[k][n] = Bt[n][k])
// Block: 128 thr (4 waves). Tile 128x64: wave w -> rows 32w..32w+31 (2 A frags),
// all 64 cols (4 B frags). 12 b128 loads : 8 WMMA per 32-K step.
// Fragments loaded straight from global with b128s, contiguous k = 16h + s in both A and B.
// ---------------------------------------------------------------------------
template <int ACT>
__global__ __launch_bounds__(128)
void gg_wmma_gemm(const unsigned short* __restrict__ A, const unsigned short* __restrict__ Bt,
                  const float* __restrict__ bias, float* __restrict__ C,
                  int M, int Nc, int Kpad) {
    const int tid   = threadIdx.x;
    const int wave  = tid >> 5;
    const int lane  = tid & 31;
    const int half  = (lane >> 4) & 1;
    const int nlane = lane & 15;
    const int rowBase = blockIdx.x * 128;
    const int colBase = blockIdx.y * 64;

    // Two A fragment source rows per wave (clamped; boundary stores guarded below)
    int arow0 = rowBase + wave * 32 + nlane;
    int arow1 = arow0 + 16;
    if (arow0 >= M) arow0 = M - 1;
    if (arow1 >= M) arow1 = M - 1;
    const unsigned short* ap0 = A + (size_t)arow0 * Kpad + 16 * half;
    const unsigned short* ap1 = A + (size_t)arow1 * Kpad + 16 * half;

    const unsigned short* bp0 = Bt + (size_t)(colBase +  0 + nlane) * Kpad + 16 * half;
    const unsigned short* bp1 = Bt + (size_t)(colBase + 16 + nlane) * Kpad + 16 * half;
    const unsigned short* bp2 = Bt + (size_t)(colBase + 32 + nlane) * Kpad + 16 * half;
    const unsigned short* bp3 = Bt + (size_t)(colBase + 48 + nlane) * Kpad + 16 * half;

    v8f acc[8];   // [mi*4 + nt]
    #pragma unroll
    for (int i = 0; i < 8; ++i) acc[i] = (v8f){0.f,0.f,0.f,0.f,0.f,0.f,0.f,0.f};

    union Frag { v16bf v; uint4 q[2]; };

    for (int k0 = 0; k0 < Kpad; k0 += 32) {
        Frag a0, a1, b0, b1, b2, b3;
        a0.q[0] = *(const uint4*)(ap0 + k0);        // K = k0+16h .. +7
        a0.q[1] = *(const uint4*)(ap0 + k0 + 8);    // K = k0+16h+8 .. +7
        a1.q[0] = *(const uint4*)(ap1 + k0);
        a1.q[1] = *(const uint4*)(ap1 + k0 + 8);
        b0.q[0] = *(const uint4*)(bp0 + k0);
        b0.q[1] = *(const uint4*)(bp0 + k0 + 8);
        b1.q[0] = *(const uint4*)(bp1 + k0);
        b1.q[1] = *(const uint4*)(bp1 + k0 + 8);
        b2.q[0] = *(const uint4*)(bp2 + k0);
        b2.q[1] = *(const uint4*)(bp2 + k0 + 8);
        b3.q[0] = *(const uint4*)(bp3 + k0);
        b3.q[1] = *(const uint4*)(bp3 + k0 + 8);

        acc[0] = __builtin_amdgcn_wmma_f32_16x16x32_bf16(false, a0.v, false, b0.v, (short)0, acc[0], false, false);
        acc[1] = __builtin_amdgcn_wmma_f32_16x16x32_bf16(false, a0.v, false, b1.v, (short)0, acc[1], false, false);
        acc[2] = __builtin_amdgcn_wmma_f32_16x16x32_bf16(false, a0.v, false, b2.v, (short)0, acc[2], false, false);
        acc[3] = __builtin_amdgcn_wmma_f32_16x16x32_bf16(false, a0.v, false, b3.v, (short)0, acc[3], false, false);
        acc[4] = __builtin_amdgcn_wmma_f32_16x16x32_bf16(false, a1.v, false, b0.v, (short)0, acc[4], false, false);
        acc[5] = __builtin_amdgcn_wmma_f32_16x16x32_bf16(false, a1.v, false, b1.v, (short)0, acc[5], false, false);
        acc[6] = __builtin_amdgcn_wmma_f32_16x16x32_bf16(false, a1.v, false, b2.v, (short)0, acc[6], false, false);
        acc[7] = __builtin_amdgcn_wmma_f32_16x16x32_bf16(false, a1.v, false, b3.v, (short)0, acc[7], false, false);
    }

    // C layout: VGPR r -> M = r + 8*half ; lane&15 -> N
    #pragma unroll
    for (int mi = 0; mi < 2; ++mi) {
        #pragma unroll
        for (int nt = 0; nt < 4; ++nt) {
            int gcol = colBase + nt * 16 + nlane;
            if (gcol >= Nc) continue;
            float bv = bias ? bias[gcol] : 0.0f;
            #pragma unroll
            for (int r = 0; r < 8; ++r) {
                int grow = rowBase + wave * 32 + mi * 16 + r + 8 * half;
                if (grow < M) {
                    float v = acc[mi * 4 + nt][r] + bv;
                    if (ACT == 1) v = fmaxf(v, 0.0f);
                    else if (ACT == 2) v = tanhf(v);
                    C[(size_t)grow * Nc + gcol] = v;
                }
            }
        }
    }
}

// ---------------------------------------------------------------------------
__global__ void gg_zero(float* p, unsigned n) {
    unsigned i = blockIdx.x * 256 + threadIdx.x;
    if (i < n) p[i] = 0.0f;
}

__global__ void gg_init_smax(unsigned* smax, int G) {
    int g = blockIdx.x * 64 + threadIdx.x;
    if (g < G) smax[g] = 0x007FFFFFu;   // encf(-inf)
}

// scatter-add: agg[dst[e]] += m[src[e]]   (float4 per work item, fp32 atomics)
__global__ void gg_scatter(const float* __restrict__ m, const int* __restrict__ src,
                           const int* __restrict__ dst, float* __restrict__ agg,
                           int E, int H) {
    unsigned idx = blockIdx.x * 256 + threadIdx.x;
    unsigned total = (unsigned)E * (unsigned)(H / 4);
    if (idx >= total) return;
    int e = idx / (H / 4);
    int f = (idx % (H / 4)) * 4;
    int s = src[e], d = dst[e];
    float4 v = *(const float4*)&m[(size_t)s * H + f];
    float* o = &agg[(size_t)d * H + f];
    __hip_atomic_fetch_add(o + 0, v.x, __ATOMIC_RELAXED, __HIP_MEMORY_SCOPE_AGENT);
    __hip_atomic_fetch_add(o + 1, v.y, __ATOMIC_RELAXED, __HIP_MEMORY_SCOPE_AGENT);
    __hip_atomic_fetch_add(o + 2, v.z, __ATOMIC_RELAXED, __HIP_MEMORY_SCOPE_AGENT);
    __hip_atomic_fetch_add(o + 3, v.w, __ATOMIC_RELAXED, __HIP_MEMORY_SCOPE_AGENT);
}

// GRU gates (torch order r,z,n), h updated in place
__global__ void gg_gru(const float* __restrict__ gi, const float* __restrict__ gh,
                       float* __restrict__ h, int N, int H) {
    unsigned idx = blockIdx.x * 256 + threadIdx.x;
    if (idx >= (unsigned)N * H) return;
    int n = idx / H, f = idx % H;
    const float* gin = gi + (size_t)n * 3 * H;
    const float* ghn = gh + (size_t)n * 3 * H;
    float r  = sigmoidf_(gin[f]         + ghn[f]);
    float z  = sigmoidf_(gin[H + f]     + ghn[H + f]);
    float nn = tanhf    (gin[2 * H + f] + r * ghn[2 * H + f]);
    float hv = h[idx];
    h[idx] = (1.0f - z) * nn + z * hv;
}

// score[n] = s1[n,:] . w2 + b2   (s1 already tanh'ed)
__global__ void gg_score(const float* __restrict__ s1, const float* __restrict__ w2,
                         const float* __restrict__ b2, float* __restrict__ score,
                         int N, int H2) {
    int n = blockIdx.x * 256 + threadIdx.x;
    if (n >= N) return;
    const float* row = s1 + (size_t)n * H2;
    float s = 0.0f;
    for (int j = 0; j < H2; ++j) s += row[j] * w2[j];
    score[n] = s + b2[0];
}

__global__ void gg_segmax(const float* __restrict__ score, const int* __restrict__ batch,
                          unsigned* __restrict__ smax, int N) {
    int n = blockIdx.x * 256 + threadIdx.x;
    if (n >= N) return;
    atomicMax(&smax[batch[n]], encf(score[n]));
}

__global__ void gg_expsum(const float* __restrict__ score, const int* __restrict__ batch,
                          const unsigned* __restrict__ smax, float* __restrict__ we,
                          float* __restrict__ denom, int N) {
    int n = blockIdx.x * 256 + threadIdx.x;
    if (n >= N) return;
    int g = batch[n];
    float e = expf(score[n] - decf(smax[g]));
    we[n] = e;
    __hip_atomic_fetch_add(&denom[g], e, __ATOMIC_RELAXED, __HIP_MEMORY_SCOPE_AGENT);
}

__global__ void gg_pool(const float* __restrict__ h, const float* __restrict__ we,
                        const float* __restrict__ denom, const int* __restrict__ batch,
                        float* __restrict__ pooled, int N, int H) {
    unsigned idx = blockIdx.x * 256 + threadIdx.x;
    if (idx >= (unsigned)N * H) return;
    int n = idx / H, f = idx % H;
    int g = batch[n];
    float w = we[n] / denom[g];
    __hip_atomic_fetch_add(&pooled[(size_t)g * H + f], h[idx] * w,
                           __ATOMIC_RELAXED, __HIP_MEMORY_SCOPE_AGENT);
}

// classifier: out[g] = relu(pooled[g] @ w1 + b1) @ w2 + b2   (block per graph)
__global__ __launch_bounds__(128)
void gg_cls(const float* __restrict__ pooled, const float* __restrict__ w1,
            const float* __restrict__ b1, const float* __restrict__ w2,
            const float* __restrict__ b2, float* __restrict__ out, int H, int H2) {
    __shared__ float red[128];
    int g = blockIdx.x, j = threadIdx.x;
    const float* p = pooled + (size_t)g * H;
    float s = b1[j];
    for (int k = 0; k < H; ++k) s += p[k] * w1[(size_t)k * H2 + j];
    s = fmaxf(s, 0.0f);
    red[j] = s * w2[j];
    __syncthreads();
    for (int st = 64; st > 0; st >>= 1) {
        if (j < st) red[j] += red[j + st];
        __syncthreads();
    }
    if (j == 0) out[g] = red[0] + b2[0];
}

// ---------------------------------------------------------------------------
extern "C" void kernel_launch(void* const* d_in, const int* in_sizes, int n_in,
                              void* d_out, int out_size, void* d_ws, size_t ws_size,
                              hipStream_t stream) {
    const float* x        = (const float*)d_in[0];
    const int*   edge     = (const int*)  d_in[1];
    const int*   batch    = (const int*)  d_in[2];
    const float* proj_w   = (const float*)d_in[3];
    const float* proj_b   = (const float*)d_in[4];
    const float* conv_w   = (const float*)d_in[5];
    const float* gru_w_ih = (const float*)d_in[6];
    const float* gru_w_hh = (const float*)d_in[7];
    const float* gru_b_ih = (const float*)d_in[8];
    const float* gru_b_hh = (const float*)d_in[9];
    const float* att_w1   = (const float*)d_in[10];
    const float* att_b1   = (const float*)d_in[11];
    const float* att_w2   = (const float*)d_in[12];
    const float* att_b2   = (const float*)d_in[13];
    const float* cls_w1   = (const float*)d_in[14];
    const float* cls_b1   = (const float*)d_in[15];
    const float* cls_w2   = (const float*)d_in[16];
    const float* cls_b2   = (const float*)d_in[17];

    const int N    = in_sizes[2];
    const int E    = in_sizes[1] / 2;
    const int H    = in_sizes[4];
    const int DIN  = in_sizes[0] / N;
    const int L    = in_sizes[5] / (H * H);
    const int G    = out_size;
    const int H2   = H / 2;
    const int Kpad = ((DIN + 31) / 32) * 32;   // 771 -> 800

    const int* srcIdx = edge;
    const int* dstIdx = edge + E;

    // ---- workspace layout ----
    float* ws  = (float*)d_ws;
    size_t NH  = (size_t)N * H;
    float* h   = ws;
    float* m   = ws + NH;
    float* agg = ws + 2 * NH;
    float* gi  = ws + 3 * NH;           // N*3H
    float* gh  = ws + 6 * NH;           // N*3H
    // bf16 buffers after the fp32 block
    unsigned short* hbf   = (unsigned short*)(ws + 9 * NH);      // N*H
    unsigned short* aggbf = hbf + NH;                            // N*H
    unsigned short* projT = aggbf + NH;                          // H*Kpad
    unsigned short* convT = projT + (size_t)H * Kpad;            // L*H*H
    unsigned short* ihbf  = convT + (size_t)L * H * H;           // 3H*H
    unsigned short* hhbf  = ihbf + (size_t)3 * H * H;            // 3H*H
    unsigned short* attT  = hhbf + (size_t)3 * H * H;            // H2*H
    // aliases (regions free at time of use)
    unsigned short* xbf = (unsigned short*)gi;   // N*Kpad ush (32MB <= 61MB region)
    float* s1     = gi;                          // N*H2
    float* score  = gh;                          // N
    float* we     = gh + N;                      // N
    float* denom  = we + N;                      // G
    float* pooled = denom + G;                   // G*H
    unsigned* smax = (unsigned*)(pooled + (size_t)G * H);

    const dim3 blk(128);
    const int mtiles = (N + 127) / 128;
    auto ctiles = [](int nc) { return (nc + 63) / 64; };
    auto eb = [](size_t n) { return (unsigned)((n + 255) / 256); };
    const int nhBlocks = (int)((NH + 255) / 256);

    // ---- 0) one-time conversions: weights (transposed to [Nc][Kpad]) + x ----
    gg_cvt_T<<<eb((size_t)H * Kpad), 256, 0, stream>>>(proj_w, projT, DIN, H, Kpad);
    for (int l = 0; l < L; ++l)
        gg_cvt_T<<<eb((size_t)H * H), 256, 0, stream>>>(
            conv_w + (size_t)l * H * H, convT + (size_t)l * H * H, H, H, H);
    gg_cvt_pad<<<eb((size_t)3 * H * H), 256, 0, stream>>>(gru_w_ih, ihbf, 3 * H, H, H);
    gg_cvt_pad<<<eb((size_t)3 * H * H), 256, 0, stream>>>(gru_w_hh, hhbf, 3 * H, H, H);
    gg_cvt_T<<<eb((size_t)H2 * H), 256, 0, stream>>>(att_w1, attT, H, H2, H);
    gg_cvt_pad<<<eb((size_t)N * Kpad), 256, 0, stream>>>(x, xbf, N, DIN, Kpad);

    // ---- 1) input projection + ReLU -> h ----
    gg_wmma_gemm<1><<<dim3(mtiles, ctiles(H)), blk, 0, stream>>>(
        xbf, projT, proj_b, h, N, H, Kpad);
    gg_cvt_pad<<<nhBlocks, 256, 0, stream>>>(h, hbf, N, H, H);

    // ---- 2) L gated message-passing layers ----
    for (int l = 0; l < L; ++l) {
        gg_wmma_gemm<0><<<dim3(mtiles, ctiles(H)), blk, 0, stream>>>(
            hbf, convT + (size_t)l * H * H, nullptr, m, N, H, H);
        gg_zero<<<nhBlocks, 256, 0, stream>>>(agg, (unsigned)NH);
        unsigned scItems = (unsigned)E * (unsigned)(H / 4);
        gg_scatter<<<(scItems + 255) / 256, 256, 0, stream>>>(m, srcIdx, dstIdx, agg, E, H);
        gg_cvt_pad<<<nhBlocks, 256, 0, stream>>>(agg, aggbf, N, H, H);
        gg_wmma_gemm<0><<<dim3(mtiles, ctiles(3 * H)), blk, 0, stream>>>(
            aggbf, ihbf, gru_b_ih, gi, N, 3 * H, H);
        gg_wmma_gemm<0><<<dim3(mtiles, ctiles(3 * H)), blk, 0, stream>>>(
            hbf, hhbf, gru_b_hh, gh, N, 3 * H, H);
        gg_gru<<<nhBlocks, 256, 0, stream>>>(gi, gh, h, N, H);
        gg_cvt_pad<<<nhBlocks, 256, 0, stream>>>(h, hbf, N, H, H);
    }

    // ---- 3) attention pooling ----
    gg_wmma_gemm<2><<<dim3(mtiles, ctiles(H2)), blk, 0, stream>>>(
        hbf, attT, att_b1, s1, N, H2, H);
    gg_score<<<(N + 255) / 256, 256, 0, stream>>>(s1, att_w2, att_b2, score, N, H2);
    gg_zero<<<eb((size_t)G + (size_t)G * H), 256, 0, stream>>>(denom, (unsigned)(G + G * H));
    gg_init_smax<<<(G + 63) / 64, 64, 0, stream>>>(smax, G);
    gg_segmax<<<(N + 255) / 256, 256, 0, stream>>>(score, batch, smax, N);
    gg_expsum<<<(N + 255) / 256, 256, 0, stream>>>(score, batch, smax, we, denom, N);
    gg_pool<<<nhBlocks, 256, 0, stream>>>(h, we, denom, batch, pooled, N, H);

    // ---- 4) classifier head -> d_out [G] ----
    gg_cls<<<G, H2, 0, stream>>>(pooled, cls_w1, cls_b1, cls_w2, cls_b2,
                                 (float*)d_out, H, H2);
}
